// CrossTransformer_43104291782756
// MI455X (gfx1250) — matrix operand order repaired and checked by reading, not compiled, not run on hardware
//
#include <hip/hip_runtime.h>
#include <hip/hip_bf16.h>

// ---------------------------------------------------------------------------
// CrossTransformer forward for MI455X (gfx1250, wave32, WMMA bf16, TDM).
// Dims: B=8 K=5 N=5 DIM=512 DK=DV=128 H=W=20 (HW=400, NKEY=2000).
// Outputs (concat): query_v [8,128,400] f32, out [8,5,128,400] f32.
// ws: wqk_bf 0 | wv_bf 131072 | qq 262144 | sk 1081344 | sv 21561344 (~42MB)
// ---------------------------------------------------------------------------

typedef __attribute__((ext_vector_type(16))) __bf16        v16bf;
typedef __attribute__((ext_vector_type(8)))  float         v8f;
typedef __attribute__((ext_vector_type(4)))  unsigned int  u32x4;
typedef __attribute__((ext_vector_type(8)))  int           i32x8;
typedef __attribute__((ext_vector_type(4)))  int           i32x4;

union BF16Frag { v16bf v; unsigned short u[16]; u32x4 q[2]; };

__device__ __forceinline__ unsigned short f2bf(float f) {
  union { __bf16 h; unsigned short u; } cv;
  cv.h = (__bf16)f;                          // native v_cvt on gfx1250
  return cv.u;
}
__device__ __forceinline__ float bf2f(unsigned short h) {
  return __uint_as_float(((unsigned int)h) << 16);
}

__device__ __forceinline__ v8f wmma_bf16(const BF16Frag& a, const BF16Frag& b, v8f c) {
  return __builtin_amdgcn_wmma_f32_16x16x32_bf16(
      false, a.v, false, b.v, (short)0, c, false, false);
}

// Fragment lane maps (16-bit operands, wave32):
//  A 16x32: lane&15 = row M; this code's k order (not AMD's table order): e -> K = 16*(lane>=16) + e
//           => one contiguous 16-ushort run at +kb16
//  B 32x16: lane&15 = col N; this code's k order (not AMD's table order): e -> K = 16*(lane>=16) + e
//           => one contiguous 16-ushort run at +kb16
//  C/D f32 16x16: VGPR r, lane l -> (M = r + 8*(l>=16), N = l&15)

// ---------------------------------------------------------------------------
// Kernel 0: convert weights to bf16 once (removes per-fragment cvt VALU).
// ---------------------------------------------------------------------------
__global__ __launch_bounds__(256) void k_cvt_weights(
    const float* __restrict__ Wqk, const float* __restrict__ Wv,
    unsigned short* __restrict__ wqk, unsigned short* __restrict__ wv) {
  const int i = blockIdx.x * 256 + threadIdx.x;      // 65536 each
  if (i < 128 * 512) { wqk[i] = f2bf(Wqk[i]); wv[i] = f2bf(Wv[i]); }
}

// ---------------------------------------------------------------------------
// Stage one 512(c) x 16(p) f32 tile (row stride 400) into LDS, then
// transpose+convert to bf16 [p][c]. TDM path on gfx1250, fallback otherwise.
// ---------------------------------------------------------------------------
__device__ __forceinline__ void stage_tile(
    const float* __restrict__ Xbase,   // &X[0*400 + p0], rows stride 400 f32
    float* sXf,                        // LDS 512*16 f32 (TDM dest)
    unsigned short* sXb,               // LDS 16*512 bf16 (transposed)
    int wv, int tid) {
#if __has_builtin(__builtin_amdgcn_tensor_load_to_lds)
  if (wv == 0) {                                     // wave-uniform: one DMA per block
    const unsigned lds_off = (unsigned)(size_t)(void*)sXf;
    const unsigned long long ga = (unsigned long long)(size_t)Xbase;
    // D# group0: count=1 | lds_addr | global_addr[56:0] | type=2
    u32x4 g0 = { 1u, lds_off, (unsigned)ga,
                 (unsigned)((ga >> 32) & 0x01FFFFFFu) | (2u << 30) };
    // D# group1: data_size=4B; tensor_dim0=400, tensor_dim1=512;
    //            tile_dim0=16, tile_dim1=512; tensor_dim0_stride=400
    i32x8 g1 = { (int)(2u << 16),          // data_size=2 (4 bytes)
                 (int)(400u << 16),        // tensor_dim0[15:0] in [31:16]
                 (int)(512u << 16),        // tensor_dim1[15:0] in [31:16]
                 (int)(16u << 16),         // tile_dim0 in [31:16]
                 512,                      // tile_dim1 (tile_dim2=0)
                 400,                      // tensor_dim0_stride[31:0]
                 0, 0 };
    i32x4 gz  = { 0, 0, 0, 0 };            // 2D: groups 2/3 unused
    i32x8 gz8 = { 0, 0, 0, 0, 0, 0, 0, 0 };
    __builtin_amdgcn_tensor_load_to_lds(g0, g1, gz, gz, gz8, 0);
    __builtin_amdgcn_s_wait_tensorcnt(0);
  }
  __syncthreads();
  // transpose + convert: sXf[c*16+p] -> sXb[p*512+c]
  for (int j = 0; j < 8; ++j) {
    const int idx = tid * 32 + j * 4;                // 4 consecutive p, same c
    const float4 x = *(const float4*)(sXf + idx);
    const int c = idx >> 4, p = idx & 15;
    sXb[(p + 0) * 512 + c] = f2bf(x.x);
    sXb[(p + 1) * 512 + c] = f2bf(x.y);
    sXb[(p + 2) * 512 + c] = f2bf(x.z);
    sXb[(p + 3) * 512 + c] = f2bf(x.w);
  }
#else
  (void)sXf;
  for (int j = 0; j < 8; ++j) {                      // coalesced fallback
    const int idx = tid * 32 + j * 4;
    const int c = idx >> 4, p = idx & 15;
    const float4 x = *(const float4*)(Xbase + c * 400 + p);
    sXb[(p + 0) * 512 + c] = f2bf(x.x);
    sXb[(p + 1) * 512 + c] = f2bf(x.y);
    sXb[(p + 2) * 512 + c] = f2bf(x.z);
    sXb[(p + 3) * 512 + c] = f2bf(x.w);
  }
#endif
  __syncthreads();
}

// ---------------------------------------------------------------------------
// Shared GEMM body: Y0 = Wqk*X, Y1 = Wv*X for one 16x16 tile (wave = M-tile).
// ---------------------------------------------------------------------------
__device__ __forceinline__ void proj_gemm(
    const unsigned short* __restrict__ wqk, const unsigned short* __restrict__ wvb,
    const unsigned short* sXb, int mt, int lo, int kb8, int kb16,
    v8f& acc0, v8f& acc1) {
  const unsigned short* A0 = wqk + (mt * 16 + lo) * 512;
  const unsigned short* A1 = wvb + (mt * 16 + lo) * 512;
  const unsigned short* Br = sXb + lo * 512;
  for (int kc = 0; kc < 16; ++kc) {
    BF16Frag a0, a1, bf;
    a0.q[0] = *(const u32x4*)(A0 + kc * 32 + kb16);
    a0.q[1] = *(const u32x4*)(A0 + kc * 32 + kb16 + 8);
    a1.q[0] = *(const u32x4*)(A1 + kc * 32 + kb16);
    a1.q[1] = *(const u32x4*)(A1 + kc * 32 + kb16 + 8);
    bf.q[0] = *(const u32x4*)(Br + kc * 32 + kb16);
    bf.q[1] = *(const u32x4*)(Br + kc * 32 + kb16 + 8);
    acc0 = wmma_bf16(a0, bf, acc0);
    acc1 = wmma_bf16(a1, bf, acc1);
  }
}

// ---------------------------------------------------------------------------
// Kernel 1: query projections. Block = (b, p-tile); waves = 8 M-tiles.
// ---------------------------------------------------------------------------
__global__ __launch_bounds__(256) void k_proj_query(
    const float* __restrict__ qrep,            // [8][512][400]
    const unsigned short* __restrict__ wqk,    // bf16 [128][512]
    const unsigned short* __restrict__ wvb,
    float* __restrict__ qv_out,                // d_out[0..409600): [8][128][400]
    unsigned short* __restrict__ qq)           // ws [8][400][128] bf16
{
  __shared__ float sXf[512 * 16];
  __shared__ unsigned short sXb[16 * 512];
  const int nt = blockIdx.x % 25, b = blockIdx.x / 25;
  const int tid = threadIdx.x;
  const int wv = tid >> 5, lane = tid & 31, lo = lane & 15, hi = lane >> 4;

  stage_tile(qrep + (size_t)b * 512 * 400 + nt * 16, sXf, sXb, wv, tid);

  v8f acc0 = {}, acc1 = {};
  proj_gemm(wqk, wvb, sXb, wv, lo, hi * 8, hi * 16, acc0, acc1);

  const int p = nt * 16 + lo;
#pragma unroll
  for (int r = 0; r < 8; ++r) {
    const int d = wv * 16 + r + hi * 8;
    qq[(b * 400 + p) * 128 + d] = f2bf(acc0[r]);
    qv_out[(b * 128 + d) * 400 + p] = acc1[r];
  }
}

// ---------------------------------------------------------------------------
// Kernel 2: support projections (dominant: 164MB read, ~21 GFLOP).
// Block = (b,k,n,p-tile); waves = 8 channel tiles; X staged via TDM.
// ---------------------------------------------------------------------------
__global__ __launch_bounds__(256) void k_proj_supports(
    const float* __restrict__ srep,            // [8][5][5][512][400]
    const unsigned short* __restrict__ wqk,
    const unsigned short* __restrict__ wvb,
    unsigned short* __restrict__ sk,           // [bk][key=2000][c=128] bf16
    unsigned short* __restrict__ sv)           // [bk][c=128][key=2000] bf16
{
  __shared__ float sXf[512 * 16];
  __shared__ unsigned short sXb[16 * 512];
  const int pt = blockIdx.x % 25;
  const int n  = (blockIdx.x / 25) % 5;
  const int k  = (blockIdx.x / 125) % 5;
  const int b  = blockIdx.x / 625;
  const int tid = threadIdx.x;
  const int wv = tid >> 5, lane = tid & 31, lo = lane & 15, hi = lane >> 4;

  stage_tile(srep + ((size_t)((b * 5 + k) * 5 + n)) * 512 * 400 + pt * 16,
             sXf, sXb, wv, tid);

  v8f acc0 = {}, acc1 = {};
  proj_gemm(wqk, wvb, sXb, wv, lo, hi * 8, hi * 16, acc0, acc1);

  const int bk  = b * 5 + k;
  const int key = n * 400 + pt * 16 + lo;
#pragma unroll
  for (int r = 0; r < 8; ++r) {
    const int d = wv * 16 + r + hi * 8;
    sk[(bk * 2000 + key) * 128 + d] = f2bf(acc0[r]);
    sv[(bk * 128 + d) * 2000 + key] = f2bf(acc1[r]);
  }
}

// ---------------------------------------------------------------------------
// Kernel 3: attention. Block per (b,k, 16-query tile); 8 waves.
// ---------------------------------------------------------------------------
__global__ __launch_bounds__(256) void k_attention(
    const unsigned short* __restrict__ qq,   // [b][p][c]
    const unsigned short* __restrict__ sk,   // [bk][key][c]
    const unsigned short* __restrict__ sv,   // [bk][c][key]
    float* __restrict__ out)                 // attn out at +409600: [bk][c][p]
{
  __shared__ unsigned short sS[16 * 2016];   // 64512 B
  __shared__ float sL[16];

  const int pt = blockIdx.x % 25;
  const int bk = blockIdx.x / 25;
  const int b  = bk / 5;
  const int tid = threadIdx.x;
  const int wv = tid >> 5, lane = tid & 31, lo = lane & 15, hi = lane >> 4;
  const int kb8 = hi * 8, kb16 = hi * 16;
  const float scale = 0.08838834764831845f;  // 128^-0.5

  // ---- Phase 1: S = scale * Q^T K -> LDS bf16 ------------------------------
  BF16Frag qa[4];
  {
    const unsigned short* Q = qq + (b * 400 + pt * 16 + lo) * 128;
#pragma unroll
    for (int cc = 0; cc < 4; ++cc) {
      qa[cc].q[0] = *(const u32x4*)(Q + cc * 32 + kb16);
      qa[cc].q[1] = *(const u32x4*)(Q + cc * 32 + kb16 + 8);
    }
  }
  for (int kt = wv; kt < 125; kt += 8) {     // wave-uniform trip counts
    v8f acc = {};
    const int key0 = kt * 16;
    const unsigned short* Kp = sk + (bk * 2000 + key0 + lo) * 128;
#pragma unroll
    for (int cc = 0; cc < 4; ++cc) {
      BF16Frag kf;
      kf.q[0] = *(const u32x4*)(Kp + cc * 32 + kb16);
      kf.q[1] = *(const u32x4*)(Kp + cc * 32 + kb16 + 8);
      acc = wmma_bf16(qa[cc], kf, acc);
    }
#pragma unroll
    for (int r = 0; r < 8; ++r)
      sS[(r + hi * 8) * 2016 + key0 + lo] = f2bf(acc[r] * scale);
  }
  __syncthreads();

  // ---- Phase 2: softmax (16 threads/row, contiguous 125-element chunks) ----
  {
    const int row = tid >> 4, sub = tid & 15;
    unsigned short* Srow = sS + row * 2016;
    float m = -3.0e38f;
    for (int j = 0; j < 125; ++j) m = fmaxf(m, bf2f(Srow[sub * 125 + j]));
    for (int o = 1; o < 16; o <<= 1) m = fmaxf(m, __shfl_xor(m, o, 16));
    float sum = 0.f;
    for (int j = 0; j < 125; ++j) {
      const int key = sub * 125 + j;
      const float e = __expf(bf2f(Srow[key]) - m);
      Srow[key] = f2bf(e);                   // unnormalized P
      sum += e;
    }
    for (int o = 1; o < 16; o <<= 1) sum += __shfl_xor(sum, o, 16);
    if (sub == 0) sL[row] = sum;
    Srow[2000 + sub] = 0;                    // zero pad cols (P=0 for K-tail)
  }
  __syncthreads();

  // ---- Phase 3: O tile (16 q x 16 c) per wave ------------------------------
  v8f acc = {};
  const int c = wv * 16 + lo;
  const unsigned short* Vrow = sv + (bk * 128 + c) * 2000;
  for (int cc = 0; cc < 63; ++cc) {          // 63*32 = 2016 (tail keys have P=0)
    BF16Frag pa, vb;
    const unsigned short* Pr = sS + lo * 2016 + cc * 32;
    pa.q[0] = *(const u32x4*)(Pr + kb16);
    pa.q[1] = *(const u32x4*)(Pr + kb16 + 8);
    const int key0 = cc * 32 + kb16;
    if (key0 + 16 <= 2000) {                 // in-range: vector loads
      vb.q[0] = *(const u32x4*)(Vrow + key0);
      vb.q[1] = *(const u32x4*)(Vrow + key0 + 8);
    } else {                                 // tail rows: P=0 anyway
      vb.q[0] = (u32x4){0, 0, 0, 0};
      vb.q[1] = (u32x4){0, 0, 0, 0};
    }
    acc = wmma_bf16(pa, vb, acc);
  }
#pragma unroll
  for (int r = 0; r < 8; ++r) {
    const int q = r + hi * 8;
    out[409600 + (bk * 128 + c) * 400 + pt * 16 + q] = acc[r] / sL[q];
  }
}

// ---------------------------------------------------------------------------
extern "C" void kernel_launch(void* const* d_in, const int* in_sizes, int n_in,
                              void* d_out, int out_size, void* d_ws, size_t ws_size,
                              hipStream_t stream) {
  (void)in_sizes; (void)n_in; (void)out_size; (void)ws_size;
  const float* qrep = (const float*)d_in[0];
  const float* srep = (const float*)d_in[1];
  const float* Wqk  = (const float*)d_in[2];
  const float* Wv   = (const float*)d_in[3];
  float* outp = (float*)d_out;

  unsigned short* wqk = (unsigned short*)d_ws;                          // 131072 B
  unsigned short* wvb = (unsigned short*)((char*)d_ws + 131072);        // 131072 B
  unsigned short* qq  = (unsigned short*)((char*)d_ws + 262144);        // 819200 B
  unsigned short* sk  = (unsigned short*)((char*)d_ws + 1081344);       // 20480000 B
  unsigned short* sv  = (unsigned short*)((char*)d_ws + 21561344);      // 20480000 B

  k_cvt_weights  <<<256,  256, 0, stream>>>(Wqk, Wv, wqk, wvb);
  k_proj_query   <<<200,  256, 0, stream>>>(qrep, wqk, wvb, outp, qq);
  k_proj_supports<<<5000, 256, 0, stream>>>(srep, wqk, wvb, sk, sv);
  k_attention    <<<1000, 256, 0, stream>>>(qq, sk, sv, outp);
}
